// FullAttentionBlock_62242666053987
// MI455X (gfx1250) — compile-verified, matrix operand order checked by reading, not run on hardware
//
#include <hip/hip_runtime.h>
#include <stdint.h>
#include <stddef.h>

typedef __attribute__((ext_vector_type(16))) __bf16 v16bf;
typedef __attribute__((ext_vector_type(8)))  float  v8f;

#define D_MODEL 1024
#define N_HEADS 16
#define HEAD_DIM 64
#define D_FF    4096
#define BATCH   4
#define SEQ     2048
#define NROWS   (BATCH * SEQ)
#define EPS     1e-6f

__device__ __forceinline__ uint16_t f2bf(float f) {
  uint32_t u = __float_as_uint(f);
  u += 0x7FFFu + ((u >> 16) & 1u);   // round-to-nearest-even
  return (uint16_t)(u >> 16);
}
__device__ __forceinline__ float bf2f(uint16_t h) {
  return __uint_as_float((uint32_t)h << 16);
}

union FragBF { v16bf v; uint32_t u[8]; };

// ---- CDNA5 async global->LDS copy (16 bytes per lane), tracked by ASYNCcnt ----
__device__ __forceinline__ void async_copy_b128(uint16_t* lds_dst, const uint16_t* g_src) {
  uint32_t loff = (uint32_t)(uintptr_t)lds_dst;  // LDS aperture: addr[31:0] = LDS offset
  asm volatile("global_load_async_to_lds_b128 %0, %1, off"
               :: "v"(loff), "v"(g_src) : "memory");
}
__device__ __forceinline__ void wait_async0() {
  asm volatile("s_wait_asynccnt 0" ::: "memory");
}

// ---------------- fp32 -> bf16 elementwise ----------------
__global__ void cvt_bf16_kernel(const float* __restrict__ in, uint16_t* __restrict__ out, size_t n) {
  size_t i = (size_t)blockIdx.x * blockDim.x + threadIdx.x;
  size_t stride = (size_t)gridDim.x * blockDim.x;
  for (; i < n; i += stride) out[i] = f2bf(in[i]);
}

// ---------------- RMSNorm: one 256-thread block per row (D=1024) ----------------
__global__ __launch_bounds__(256) void rmsnorm_bf16_kernel(const float* __restrict__ x,
                                                           const float* __restrict__ g,
                                                           uint16_t* __restrict__ out) {
  __shared__ float red[256];
  const int row = blockIdx.x;
  const float* xr = x + (size_t)row * D_MODEL;
  float v[4]; float s = 0.f;
#pragma unroll
  for (int i = 0; i < 4; ++i) { v[i] = xr[threadIdx.x + i * 256]; s += v[i] * v[i]; }
  red[threadIdx.x] = s;
  __syncthreads();
  for (int off = 128; off > 0; off >>= 1) {
    if (threadIdx.x < off) red[threadIdx.x] += red[threadIdx.x + off];
    __syncthreads();
  }
  const float inv = rsqrtf(red[0] * (1.f / D_MODEL) + EPS);
  uint16_t* o = out + (size_t)row * D_MODEL;
#pragma unroll
  for (int i = 0; i < 4; ++i) {
    int c = threadIdx.x + i * 256;
    o[c] = f2bf(v[i] * inv * g[c]);
  }
}

// ---------------- RoPE + permute [B,S,H*64] -> [B,H,S,64] ----------------
__global__ void rope_permute_kernel(const uint16_t* __restrict__ in,
                                    const float* __restrict__ cosb,
                                    const float* __restrict__ sinb,
                                    uint16_t* __restrict__ out) {
  size_t idx = (size_t)blockIdx.x * blockDim.x + threadIdx.x;
  const size_t total = (size_t)BATCH * N_HEADS * SEQ * HEAD_DIM;
  const size_t stride = (size_t)gridDim.x * blockDim.x;
  for (; idx < total; idx += stride) {
    int d = (int)(idx & 63);
    size_t tt = idx >> 6;
    int s = (int)(tt % SEQ); tt /= SEQ;
    int h = (int)(tt % N_HEADS);
    int b = (int)(tt / N_HEADS);
    size_t ib = ((size_t)(b * SEQ + s)) * D_MODEL + h * HEAD_DIM;
    float xv = bf2f(in[ib + d]);
    float ot = bf2f(in[ib + (d ^ 32)]);   // rotate_half partner
    float c = cosb[s * HEAD_DIM + d], sn = sinb[s * HEAD_DIM + d];
    float rot = (d < 32) ? -ot : ot;
    out[idx] = f2bf(xv * c + rot * sn);
  }
}

// ---------------- V permute [B,S,H*64] -> [B,H,64,S] (contraction-contiguous) ----------------
__global__ void v_transpose_kernel(const uint16_t* __restrict__ in, uint16_t* __restrict__ out) {
  size_t idx = (size_t)blockIdx.x * blockDim.x + threadIdx.x;
  const size_t total = (size_t)BATCH * N_HEADS * HEAD_DIM * SEQ;
  const size_t stride = (size_t)gridDim.x * blockDim.x;
  for (; idx < total; idx += stride) {
    int s = (int)(idx & (SEQ - 1));
    size_t tt = idx >> 11;
    int d = (int)(tt & 63); tt >>= 6;
    int h = (int)(tt & (N_HEADS - 1));
    int b = (int)(tt >> 4);
    out[idx] = in[((size_t)(b * SEQ + s)) * D_MODEL + h * HEAD_DIM + d];
  }
}

// ---------------- bf16 WMMA GEMM: C[M,N] = A[M,K] x B[K,N] ----------------
// Block tile 128x64, BK=32. 8 waves, each owns a 32x32 quad (2x2 WMMA tiles).
// A staged via global_load_async_to_lds_b128; B staged transposed via DS stores.
// EPI 0: store bf16.  EPI 1: store f32 = res + acc.  EPI 2: store bf16 = silu(acc) * aux.
template <int EPI>
__global__ __launch_bounds__(256)
void gemm_bf16_kernel(const uint16_t* __restrict__ A, const uint16_t* __restrict__ B,
                      void* __restrict__ C, const float* __restrict__ res,
                      const uint16_t* __restrict__ aux, int M, int N, int K) {
  __shared__ uint16_t lA[128 * 32];   // [m][k]
  __shared__ uint16_t lBt[64 * 32];   // [n][k]  (B transposed in LDS)
  const int n0 = blockIdx.x * 64;
  const int m0 = blockIdx.y * 128;
  const int t = threadIdx.x;
  const int lane = t & 31;
  const int wave = t >> 5;
  const int mq = wave & 3;     // wave's 32-row M quad (0..3)
  const int nq = wave >> 2;    // wave's 32-col N quad (0..1)
  const int half = lane >> 4, lr = lane & 15;

  v8f zero = {};
  v8f acc[2][2] = {{zero, zero}, {zero, zero}};

  const int arow = t >> 1, acol = (t & 1) * 16;  // A: 128 rows x 32 cols, 16 elems/thread
  const int brow = t >> 3, bcol = (t & 7) * 8;   // B: 32 rows x 64 cols,  8 elems/thread

  for (int k0 = 0; k0 < K; k0 += 32) {
    // A tile: two async 16B lane-copies straight into LDS (ASYNCcnt path)
    const uint16_t* ga = &A[(size_t)(m0 + arow) * K + k0 + acol];
    async_copy_b128(&lA[arow * 32 + acol], ga);
    async_copy_b128(&lA[arow * 32 + acol + 8], ga + 8);
    // B tile: vector load + transposing scatter into LDS
    union { uint4 v; uint16_t h[8]; } bv;
    bv.v = *(const uint4*)&B[(size_t)(k0 + brow) * N + n0 + bcol];
#pragma unroll
    for (int j = 0; j < 8; ++j) lBt[(bcol + j) * 32 + brow] = bv.h[j];
    if (k0 + 32 < K) {
      __builtin_prefetch(&A[(size_t)(m0 + arow) * K + k0 + 32 + acol], 0, 1);
      __builtin_prefetch(&B[(size_t)(k0 + 32 + brow) * N + n0 + bcol], 0, 1);
    }
    wait_async0();
    __syncthreads();

    FragBF fa[2], fb[2];
#pragma unroll
    for (int j = 0; j < 2; ++j) {
      const uint16_t* ap = &lA[(mq * 32 + j * 16 + lr) * 32];
#pragma unroll
      for (int r = 0; r < 8; ++r) {
        int kk = ((r >> 2) << 4) + (half << 3) + ((r & 3) << 1);
        fa[j].u[r] = *(const uint32_t*)&ap[kk];
      }
    }
#pragma unroll
    for (int i = 0; i < 2; ++i) {
      const uint16_t* bp = &lBt[(nq * 32 + i * 16 + lr) * 32];
#pragma unroll
      for (int r = 0; r < 8; ++r) {
        int kk = ((r >> 2) << 4) + (half << 3) + ((r & 3) << 1);
        fb[i].u[r] = *(const uint32_t*)&bp[kk];
      }
    }
#pragma unroll
    for (int j = 0; j < 2; ++j)
#pragma unroll
      for (int i = 0; i < 2; ++i)
        acc[j][i] = __builtin_amdgcn_wmma_f32_16x16x32_bf16(false, fa[j].v, false, fb[i].v,
                                                            (short)0, acc[j][i], false, false);
    __syncthreads();
  }

#pragma unroll
  for (int j = 0; j < 2; ++j)
#pragma unroll
    for (int i = 0; i < 2; ++i) {
      int n = n0 + nq * 32 + i * 16 + lr;
#pragma unroll
      for (int r = 0; r < 8; ++r) {
        int m = m0 + mq * 32 + j * 16 + r + 8 * half;
        size_t idx = (size_t)m * N + n;
        float val = acc[j][i][r];
        if (EPI == 0) {
          ((uint16_t*)C)[idx] = f2bf(val);
        } else if (EPI == 1) {
          ((float*)C)[idx] = res[idx] + val;
        } else {
          float sg = val / (1.f + __expf(-val));           // silu
          ((uint16_t*)C)[idx] = f2bf(sg * bf2f(aux[idx])); // * up
        }
      }
    }
}

// ---------------- Flash attention: 1 wave per 16-query tile, 32-key tiles ----------------
__global__ __launch_bounds__(32)
void flash_attn_kernel(const uint16_t* __restrict__ Q,   // [B,H,S,64]
                       const uint16_t* __restrict__ Kd,  // [B,H,S,64]
                       const uint16_t* __restrict__ Vt,  // [B,H,64,S]
                       uint16_t* __restrict__ ctx) {     // [B,S,D]
  __shared__ uint16_t lP[16 * 32];
  const int q0 = blockIdx.x * 16;
  const int h = blockIdx.y;
  const int b = blockIdx.z;
  const int lane = threadIdx.x;
  const int half = lane >> 4, lr = lane & 15;
  const size_t bh = (size_t)(b * N_HEADS + h);
  const uint16_t* qb = Q + bh * SEQ * HEAD_DIM;
  const uint16_t* kb = Kd + bh * SEQ * HEAD_DIM;
  const uint16_t* vb = Vt + bh * HEAD_DIM * SEQ;

  FragBF fq[2];  // Q 16x64 as two 16x32 A-fragments
#pragma unroll
  for (int kh = 0; kh < 2; ++kh)
#pragma unroll
    for (int r = 0; r < 8; ++r) {
      int kk = kh * 32 + ((r >> 2) << 4) + (half << 3) + ((r & 3) << 1);
      fq[kh].u[r] = *(const uint32_t*)&qb[(size_t)(q0 + lr) * 64 + kk];
    }

  v8f zero = {};
  v8f o[4] = {zero, zero, zero, zero};
  float mrow[8], lrow[8];
#pragma unroll
  for (int r = 0; r < 8; ++r) { mrow[r] = -3.0e38f; lrow[r] = 0.f; }

  const int ntiles = (q0 + 16 + 31) / 32;
  for (int kt = 0; kt < ntiles; ++kt) {
    const int kc = kt * 32;
    v8f sc[2] = {zero, zero};
#pragma unroll
    for (int nt = 0; nt < 2; ++nt)
#pragma unroll
      for (int kh = 0; kh < 2; ++kh) {
        FragBF fk;
#pragma unroll
        for (int r = 0; r < 8; ++r) {
          int kk = kh * 32 + ((r >> 2) << 4) + (half << 3) + ((r & 3) << 1);
          fk.u[r] = *(const uint32_t*)&kb[(size_t)(kc + nt * 16 + lr) * 64 + kk];
        }
        sc[nt] = __builtin_amdgcn_wmma_f32_16x16x32_bf16(false, fq[kh].v, false, fk.v,
                                                         (short)0, sc[nt], false, false);
      }

    float ss[2][8];
#pragma unroll
    for (int nt = 0; nt < 2; ++nt)
#pragma unroll
      for (int r = 0; r < 8; ++r) {
        int qrow = q0 + r + 8 * half;
        int key = kc + nt * 16 + lr;
        float sval = sc[nt][r] * 0.125f;  // 1/sqrt(64)
        ss[nt][r] = (key > qrow) ? -1.0e30f : sval;
      }

#pragma unroll
    for (int r = 0; r < 8; ++r) {
      float mx = fmaxf(ss[0][r], ss[1][r]);
      mx = fmaxf(mx, __shfl_xor(mx, 1, 32));
      mx = fmaxf(mx, __shfl_xor(mx, 2, 32));
      mx = fmaxf(mx, __shfl_xor(mx, 4, 32));
      mx = fmaxf(mx, __shfl_xor(mx, 8, 32));   // max within 16-lane half-group
      float nm = fmaxf(mrow[r], mx);
      float al = __expf(mrow[r] - nm);
      float p0 = __expf(ss[0][r] - nm);
      float p1 = __expf(ss[1][r] - nm);
      float sum = p0 + p1;
      sum += __shfl_xor(sum, 1, 32);
      sum += __shfl_xor(sum, 2, 32);
      sum += __shfl_xor(sum, 4, 32);
      sum += __shfl_xor(sum, 8, 32);
      lrow[r] = lrow[r] * al + sum;
      mrow[r] = nm;
#pragma unroll
      for (int j = 0; j < 4; ++j) o[j][r] *= al;
      int m = r + 8 * half;
      lP[m * 32 + lr] = f2bf(p0);
      lP[m * 32 + 16 + lr] = f2bf(p1);
    }
    __syncthreads();

    FragBF fp;  // P 16x32 in A layout
#pragma unroll
    for (int r = 0; r < 8; ++r) {
      int kk = ((r >> 2) << 4) + (half << 3) + ((r & 3) << 1);
      fp.u[r] = *(const uint32_t*)&lP[lr * 32 + kk];
    }
#pragma unroll
    for (int j = 0; j < 4; ++j) {
      FragBF fv;  // V 32x16 B-frag from [d][s] layout: contiguous along k
#pragma unroll
      for (int r = 0; r < 8; ++r) {
        int kk = ((r >> 2) << 4) + (half << 3) + ((r & 3) << 1);
        fv.u[r] = *(const uint32_t*)&vb[(size_t)(j * 16 + lr) * SEQ + kc + kk];
      }
      o[j] = __builtin_amdgcn_wmma_f32_16x16x32_bf16(false, fp.v, false, fv.v,
                                                     (short)0, o[j], false, false);
    }
    __syncthreads();
  }

#pragma unroll
  for (int j = 0; j < 4; ++j) {
    int d = j * 16 + lr;
#pragma unroll
    for (int r = 0; r < 8; ++r) {
      int m = r + 8 * half;
      float val = o[j][r] / lrow[r];
      size_t idx = ((size_t)(b * SEQ + q0 + m)) * D_MODEL + h * HEAD_DIM + d;
      ctx[idx] = f2bf(val);
    }
  }
}

// ---------------- host launcher ----------------
extern "C" void kernel_launch(void* const* d_in, const int* in_sizes, int n_in,
                              void* d_out, int out_size, void* d_ws, size_t ws_size,
                              hipStream_t stream) {
  const float* x  = (const float*)d_in[0];
  const float* rc = (const float*)d_in[1];
  const float* rs = (const float*)d_in[2];
  const float* g1 = (const float*)d_in[3];
  const float* g2 = (const float*)d_in[4];
  const float* wq = (const float*)d_in[5];
  const float* wk = (const float*)d_in[6];
  const float* wv = (const float*)d_in[7];
  const float* wo = (const float*)d_in[8];
  const float* wg = (const float*)d_in[9];
  const float* wu = (const float*)d_in[10];
  const float* wd = (const float*)d_in[11];

  char* cur = (char*)d_ws;
  auto alloc = [&](size_t bytes) {
    char* p = cur;
    cur += (bytes + 255) & ~(size_t)255;
    return p;
  };
  const size_t NB = (size_t)NROWS * D_MODEL;      // 8M
  const size_t NF = (size_t)NROWS * D_FF;         // 33.5M
  const size_t WDD = (size_t)D_MODEL * D_MODEL;   // 1M
  const size_t WDF = (size_t)D_MODEL * D_FF;      // 4M

  uint16_t* xn1 = (uint16_t*)alloc(NB * 2);
  uint16_t* wqb = (uint16_t*)alloc(WDD * 2);
  uint16_t* wkb = (uint16_t*)alloc(WDD * 2);
  uint16_t* wvb = (uint16_t*)alloc(WDD * 2);
  uint16_t* wob = (uint16_t*)alloc(WDD * 2);
  uint16_t* wgb = (uint16_t*)alloc(WDF * 2);
  uint16_t* wub = (uint16_t*)alloc(WDF * 2);
  uint16_t* wdb = (uint16_t*)alloc(WDF * 2);
  uint16_t* qb  = (uint16_t*)alloc(NB * 2);
  uint16_t* kbf = (uint16_t*)alloc(NB * 2);
  uint16_t* vbf = (uint16_t*)alloc(NB * 2);
  uint16_t* qr  = (uint16_t*)alloc(NB * 2);
  uint16_t* kr  = (uint16_t*)alloc(NB * 2);
  uint16_t* vt  = (uint16_t*)alloc(NB * 2);
  uint16_t* ctx = (uint16_t*)alloc(NB * 2);
  float*    x1  = (float*)alloc(NB * 4);
  uint16_t* xn2 = (uint16_t*)alloc(NB * 2);
  uint16_t* up  = (uint16_t*)alloc(NF * 2);
  uint16_t* hbf = (uint16_t*)alloc(NF * 2);

  auto cvt = [&](const float* src, uint16_t* dst, size_t n) {
    int blocks = (int)((n + 255) / 256);
    cvt_bf16_kernel<<<blocks, 256, 0, stream>>>(src, dst, n);
  };
  cvt(wq, wqb, WDD); cvt(wk, wkb, WDD); cvt(wv, wvb, WDD); cvt(wo, wob, WDD);
  cvt(wg, wgb, WDF); cvt(wu, wub, WDF); cvt(wd, wdb, WDF);

  // xn1 = rmsnorm(x, g1)
  rmsnorm_bf16_kernel<<<NROWS, 256, 0, stream>>>(x, g1, xn1);

  // Q/K/V projections
  dim3 gProj(D_MODEL / 64, NROWS / 128);
  gemm_bf16_kernel<0><<<gProj, 256, 0, stream>>>(xn1, wqb, qb, nullptr, nullptr, NROWS, D_MODEL, D_MODEL);
  gemm_bf16_kernel<0><<<gProj, 256, 0, stream>>>(xn1, wkb, kbf, nullptr, nullptr, NROWS, D_MODEL, D_MODEL);
  gemm_bf16_kernel<0><<<gProj, 256, 0, stream>>>(xn1, wvb, vbf, nullptr, nullptr, NROWS, D_MODEL, D_MODEL);

  // RoPE + layout permutes
  int permBlocks = (int)((NB + 255) / 256);
  rope_permute_kernel<<<permBlocks, 256, 0, stream>>>(qb, rc, rs, qr);
  rope_permute_kernel<<<permBlocks, 256, 0, stream>>>(kbf, rc, rs, kr);
  v_transpose_kernel<<<permBlocks, 256, 0, stream>>>(vbf, vt);

  // causal flash attention
  dim3 gAtt(SEQ / 16, N_HEADS, BATCH);
  flash_attn_kernel<<<gAtt, 32, 0, stream>>>(qr, kr, vt, ctx);

  // x1 = x + ctx @ wo
  gemm_bf16_kernel<1><<<gProj, 256, 0, stream>>>(ctx, wob, x1, x, nullptr, NROWS, D_MODEL, D_MODEL);

  // xn2 = rmsnorm(x1, g2)
  rmsnorm_bf16_kernel<<<NROWS, 256, 0, stream>>>(x1, g2, xn2);

  // up = xn2 @ w_up ; h = silu(xn2 @ w_gate) * up
  dim3 gFF(D_FF / 64, NROWS / 128);
  gemm_bf16_kernel<0><<<gFF, 256, 0, stream>>>(xn2, wub, up, nullptr, nullptr, NROWS, D_FF, D_MODEL);
  gemm_bf16_kernel<2><<<gFF, 256, 0, stream>>>(xn2, wgb, hbf, nullptr, up, NROWS, D_FF, D_MODEL);

  // out = x1 + h @ w_down
  gemm_bf16_kernel<1><<<gProj, 256, 0, stream>>>(hbf, wdb, (float*)d_out, x1, nullptr, NROWS, D_MODEL, D_FF);
}
